// NomicBertAttention_71030169141255
// MI455X (gfx1250) — matrix operand order repaired and checked by reading, not compiled, not run on hardware
//
#include <hip/hip_runtime.h>
#include <hip/hip_bf16.h>

typedef __bf16 bf16;
typedef __attribute__((ext_vector_type(16))) __bf16 v16bf;
typedef __attribute__((ext_vector_type(8)))  __bf16 v8bf;
typedef __attribute__((ext_vector_type(4)))  __bf16 v4bf;
typedef __attribute__((ext_vector_type(8)))  float  v8f;
typedef __attribute__((ext_vector_type(4)))  float  v4f;

#define BATCH 2
#define SEQ   2048
#define DIM   1024
#define HEADS 16
#define HDIM  64
#define ROWS  (BATCH*SEQ)          // 4096
#define SCORE_SCALE 0.015625f      // SCALING^2 = 1/64

// ---------------- WMMA helpers ----------------

__device__ __forceinline__ v8f wmma_bf16(v16bf a, v16bf b, v8f c) {
  return __builtin_amdgcn_wmma_f32_16x16x32_bf16(
      /*neg_a=*/false, a, /*neg_b=*/false, b,
      /*c_mod=*/(short)0, c, /*reuse_a=*/false, /*reuse_b=*/false);
}

// A-matrix fragment (16x32 bf16). Lane L (0-15) holds row L,
// K={0..7} in v0..3, K={16..23} in v4..7; lanes 16-31 hold K={8..15,24..31}.
__device__ __forceinline__ v16bf load_fragA(const bf16* p, int lane) {
  const int off = (lane & 16) ? 8 : 0;
  v8bf lo = *(const v8bf*)(p + off);
  v8bf hi = *(const v8bf*)(p + off + 16);
  v16bf r;
#pragma unroll
  for (int i = 0; i < 8; ++i) { r[i] = lo[i]; r[i + 8] = hi[i]; }
  return r;
}

// B-matrix fragment (32x16 bf16). Lanes 0-15 hold K={0..7,16..23},
// lanes 16-31 hold K={8..15,24..31}. p = this lane's column data (contiguous K).
__device__ __forceinline__ v16bf load_fragB(const bf16* p, int lane) {
  const int off = (lane & 16) ? 8 : 0;
  v8bf lo = *(const v8bf*)(p + off);
  v8bf hi = *(const v8bf*)(p + off + 16);
  v16bf r;
#pragma unroll
  for (int i = 0; i < 8; ++i) { r[i] = lo[i]; r[i + 8] = hi[i]; }
  return r;
}

__device__ __forceinline__ float rmax16(float v) {
  v = fmaxf(v, __shfl_xor(v, 1));
  v = fmaxf(v, __shfl_xor(v, 2));
  v = fmaxf(v, __shfl_xor(v, 4));
  v = fmaxf(v, __shfl_xor(v, 8));
  return v;
}
__device__ __forceinline__ float rsum16(float v) {
  v += __shfl_xor(v, 1);
  v += __shfl_xor(v, 2);
  v += __shfl_xor(v, 4);
  v += __shfl_xor(v, 8);
  return v;
}

// ---------------- Kernel 1: f32 -> bf16 conversion (vectorized x4) ----------------
__global__ void cvt_kernel(const float* __restrict__ hs,
                           const float* __restrict__ wq, const float* __restrict__ wk,
                           const float* __restrict__ wv, const float* __restrict__ wo,
                           bf16* __restrict__ xb,
                           bf16* __restrict__ wqb, bf16* __restrict__ wkb,
                           bf16* __restrict__ wvb, bf16* __restrict__ wob) {
  int idx = blockIdx.x * blockDim.x + threadIdx.x;   // (8M/4) threads
  int e = idx * 4;
  const float* src;
  bf16* dst;
  int off;
  if (e < ROWS * DIM) {
    src = hs; dst = xb; off = e;
  } else {
    int r = e - ROWS * DIM;
    int wsel = r >> 20;
    off = r & (1048576 - 1);
    src = (wsel == 0) ? wq : (wsel == 1) ? wk : (wsel == 2) ? wv : wo;
    dst = (wsel == 0) ? wqb : (wsel == 1) ? wkb : (wsel == 2) ? wvb : wob;
  }
  v4f v = *(const v4f*)(src + off);
  v4bf o;
#pragma unroll
  for (int i = 0; i < 4; ++i) o[i] = (bf16)v[i];
  *(v4bf*)(dst + off) = o;
}

// ---------------- Kernel 2: QKV projection GEMM ----------------
// C[m,n] = sum_k X[m,k]*W[n,k] + b[n]. Wave computes 32x64 tile (2 M-tiles x
// 4 N-tiles), software-pipelined: next-k fragments load while current-k WMMAs run.
// mode 0 -> Q [B,H,S,64], mode 1 -> K [B,H,S,64], mode 2 -> V^T [B,H,64,S].
__global__ void qkv_gemm_kernel(const bf16* __restrict__ X,
                                const bf16* __restrict__ Wqb, const bf16* __restrict__ Wkb,
                                const bf16* __restrict__ Wvb,
                                const float* __restrict__ bq, const float* __restrict__ bk,
                                const float* __restrict__ bv,
                                bf16* __restrict__ Qb, bf16* __restrict__ Kb,
                                bf16* __restrict__ Vt) {
  const int mode = blockIdx.y;
  const bf16* W = (mode == 0) ? Wqb : (mode == 1) ? Wkb : Wvb;
  const float* bias = (mode == 0) ? bq : (mode == 1) ? bk : bv;

  const int lane = threadIdx.x & 31, ln = lane & 15, half = lane >> 4;
  const int wid = blockIdx.x * 8 + (threadIdx.x >> 5);
  const int m0 = (wid >> 4) * 32;     // 128 M-tiles of 32 rows
  const int n0 = (wid & 15) * 64;     // 16 N-strips of 64 cols

  const bf16* arow0 = X + (m0 + ln) * DIM;
  const bf16* arow1 = X + (m0 + 16 + ln) * DIM;
  const bf16* wrow[4];
#pragma unroll
  for (int j = 0; j < 4; ++j) wrow[j] = W + (n0 + j * 16 + ln) * DIM;

  v8f acc[2][4] = {};
  // prologue loads (k=0)
  v16bf a0 = load_fragA(arow0, lane);
  v16bf a1 = load_fragA(arow1, lane);
  v16bf b[4];
#pragma unroll
  for (int j = 0; j < 4; ++j) b[j] = load_fragB(wrow[j], lane);

  for (int k0 = 0; k0 < DIM; k0 += 32) {
    // issue next iteration's loads first (wraps to 0 on last iter; harmless)
    int kn = (k0 + 32 < DIM) ? (k0 + 32) : 0;
    v16bf a0n = load_fragA(arow0 + kn, lane);
    v16bf a1n = load_fragA(arow1 + kn, lane);
    v16bf bn[4];
#pragma unroll
    for (int j = 0; j < 4; ++j) bn[j] = load_fragB(wrow[j] + kn, lane);

#pragma unroll
    for (int j = 0; j < 4; ++j) {
      acc[0][j] = wmma_bf16(a0, b[j], acc[0][j]);
      acc[1][j] = wmma_bf16(a1, b[j], acc[1][j]);
    }
    a0 = a0n; a1 = a1n;
#pragma unroll
    for (int j = 0; j < 4; ++j) b[j] = bn[j];
  }

#pragma unroll
  for (int mi = 0; mi < 2; ++mi) {
#pragma unroll
    for (int j = 0; j < 4; ++j) {
      int n = n0 + j * 16 + ln;
      float bval = bias[n];
      int hcol = n >> 6, hd = n & 63;
#pragma unroll
      for (int i = 0; i < 8; ++i) {
        int m = m0 + mi * 16 + i + 8 * half;
        int bb = m >> 11, s = m & (SEQ - 1);
        float v = acc[mi][j][i] + bval;
        if (mode == 0)      Qb[((bb * HEADS + hcol) * SEQ + s) * HDIM + hd] = (bf16)v;
        else if (mode == 1) Kb[((bb * HEADS + hcol) * SEQ + s) * HDIM + hd] = (bf16)v;
        else                Vt[((bb * HEADS + hcol) * HDIM + hd) * SEQ + s] = (bf16)v;
      }
    }
  }
}

// ---------------- Kernel 3: RoPE (in place on Q,K) ----------------
__global__ void rope_kernel(bf16* __restrict__ Q, bf16* __restrict__ K,
                            const float* __restrict__ cosp, const float* __restrict__ sinp) {
  int idx = blockIdx.x * blockDim.x + threadIdx.x;   // B*H*S*32 threads
  int d = idx & 31;
  int s = (idx >> 5) & (SEQ - 1);
  int bh = idx >> 16;
  float c = cosp[s * HDIM + d];
  float sn = sinp[s * HDIM + d];
  int base = (bh * SEQ + s) * HDIM;
  float q1 = (float)Q[base + d], q2 = (float)Q[base + d + 32];
  Q[base + d]      = (bf16)(q1 * c - q2 * sn);
  Q[base + d + 32] = (bf16)(q2 * c + q1 * sn);
  float k1 = (float)K[base + d], k2 = (float)K[base + d + 32];
  K[base + d]      = (bf16)(k1 * c - k2 * sn);
  K[base + d + 32] = (bf16)(k2 * c + k1 * sn);
}

// ---------------- Kernel 4: flash attention ----------------
// One wave = one 16-query tile of one (b,h). 32 keys per iteration, 8 WMMA/iter.
__global__ void attn_kernel(const bf16* __restrict__ Qb, const bf16* __restrict__ Kb,
                            const bf16* __restrict__ Vt, bf16* __restrict__ Ctx) {
  __shared__ bf16 plds[4][2][16 * 32];   // per-wave double-buffered P staging
  const int warp = threadIdx.x >> 5;
  const int lane = threadIdx.x & 31, ln = lane & 15, half = lane >> 4;
  const int wid = blockIdx.x * 4 + warp;
  const int qt = wid & 127;            // S/16 = 128 query tiles
  const int bh = wid >> 7;             // 32 (b,h) pairs
  const int q0 = qt * 16;

  const bf16* qrow = Qb + ((bh * SEQ) + q0 + ln) * HDIM;
  v16bf aq0 = load_fragA(qrow, lane);
  v16bf aq1 = load_fragA(qrow + 32, lane);

  float mi[8], li[8];
#pragma unroll
  for (int i = 0; i < 8; ++i) { mi[i] = -1e30f; li[i] = 0.f; }
  v8f acc[4] = {};

  const bf16* kbase = Kb + bh * SEQ * HDIM;
  const bf16* vbase = Vt + bh * HDIM * SEQ;

  for (int t = 0; t < SEQ; t += 32) {
    // ---- scores S = Q K^T * (1/64), two 16-key tiles ----
    const bf16* k0p = kbase + (t + ln) * HDIM;
    const bf16* k1p = kbase + (t + 16 + ln) * HDIM;
    v8f s0 = {}, s1 = {};
    s0 = wmma_bf16(aq0, load_fragB(k0p, lane), s0);
    s0 = wmma_bf16(aq1, load_fragB(k0p + 32, lane), s0);
    s1 = wmma_bf16(aq0, load_fragB(k1p, lane), s1);
    s1 = wmma_bf16(aq1, load_fragB(k1p + 32, lane), s1);

    // prefetch the K/V stream 64 keys ahead (global_prefetch_b8)
    if (t + 64 < SEQ) {
      __builtin_prefetch(kbase + (t + 64 + ln) * HDIM, 0, 3);
      __builtin_prefetch(kbase + (t + 80 + ln) * HDIM, 0, 3);
#pragma unroll
      for (int j = 0; j < 4; ++j)
        __builtin_prefetch(vbase + (j * 16 + ln) * SEQ + t + 64, 0, 3);
    }

    // ---- online softmax ----
    float mn[8], alpha[8];
#pragma unroll
    for (int i = 0; i < 8; ++i) {
      float a = s0[i] * SCORE_SCALE;
      float b = s1[i] * SCORE_SCALE;
      s0[i] = a; s1[i] = b;
      float tm = rmax16(fmaxf(a, b));
      float m2 = fmaxf(mi[i], tm);
      alpha[i] = __expf(mi[i] - m2);
      mn[i] = m2;
    }
    bf16* pbuf = &plds[warp][(t >> 5) & 1][0];
#pragma unroll
    for (int i = 0; i < 8; ++i) {
      float p0 = __expf(s0[i] - mn[i]);
      float p1 = __expf(s1[i] - mn[i]);
      li[i] = li[i] * alpha[i] + rsum16(p0 + p1);
      mi[i] = mn[i];
      int r = i + 8 * half;
      pbuf[r * 32 + ln]      = (bf16)p0;   // keys t..t+15
      pbuf[r * 32 + 16 + ln] = (bf16)p1;   // keys t+16..t+31
#pragma unroll
      for (int j = 0; j < 4; ++j) acc[j][i] *= alpha[i];
    }
    // intra-wave cross-lane LDS visibility
    asm volatile("s_wait_dscnt 0x0" ::: "memory");
    v16bf pa = load_fragA(pbuf + ln * 32, lane);   // P as A-fragment (16x32)
    asm volatile("" ::: "memory");

    // ---- ctx += P * V, V^T rows contiguous along keys ----
#pragma unroll
    for (int j = 0; j < 4; ++j) {
      v16bf bv = load_fragB(vbase + (j * 16 + ln) * SEQ + t, lane);
      acc[j] = wmma_bf16(pa, bv, acc[j]);
    }
  }

  // ---- normalize and write ctx [B,S,D] as bf16 ----
  float inv[8];
#pragma unroll
  for (int i = 0; i < 8; ++i) inv[i] = 1.0f / li[i];
  const int b = bh >> 4, h = bh & 15;
#pragma unroll
  for (int j = 0; j < 4; ++j) {
    int col = h * HDIM + j * 16 + ln;
#pragma unroll
    for (int i = 0; i < 8; ++i) {
      int srow = q0 + i + 8 * half;
      Ctx[(b * SEQ + srow) * DIM + col] = (bf16)(acc[j][i] * inv[i]);
    }
  }
}

// ---------------- Kernel 5: output projection + bias + residual ----------------
__global__ void oproj_gemm_kernel(const bf16* __restrict__ Ctx, const bf16* __restrict__ Wob,
                                  const float* __restrict__ bo, const float* __restrict__ hs,
                                  float* __restrict__ out) {
  const int lane = threadIdx.x & 31, ln = lane & 15, half = lane >> 4;
  const int wid = blockIdx.x * 8 + (threadIdx.x >> 5);
  const int m0 = (wid >> 4) * 32;
  const int n0 = (wid & 15) * 64;

  const bf16* arow0 = Ctx + (m0 + ln) * DIM;
  const bf16* arow1 = Ctx + (m0 + 16 + ln) * DIM;
  const bf16* wrow[4];
#pragma unroll
  for (int j = 0; j < 4; ++j) wrow[j] = Wob + (n0 + j * 16 + ln) * DIM;

  v8f acc[2][4] = {};
  v16bf a0 = load_fragA(arow0, lane);
  v16bf a1 = load_fragA(arow1, lane);
  v16bf b[4];
#pragma unroll
  for (int j = 0; j < 4; ++j) b[j] = load_fragB(wrow[j], lane);

  for (int k0 = 0; k0 < DIM; k0 += 32) {
    int kn = (k0 + 32 < DIM) ? (k0 + 32) : 0;
    v16bf a0n = load_fragA(arow0 + kn, lane);
    v16bf a1n = load_fragA(arow1 + kn, lane);
    v16bf bn[4];
#pragma unroll
    for (int j = 0; j < 4; ++j) bn[j] = load_fragB(wrow[j] + kn, lane);

#pragma unroll
    for (int j = 0; j < 4; ++j) {
      acc[0][j] = wmma_bf16(a0, b[j], acc[0][j]);
      acc[1][j] = wmma_bf16(a1, b[j], acc[1][j]);
    }
    a0 = a0n; a1 = a1n;
#pragma unroll
    for (int j = 0; j < 4; ++j) b[j] = bn[j];
  }

#pragma unroll
  for (int mi = 0; mi < 2; ++mi) {
#pragma unroll
    for (int j = 0; j < 4; ++j) {
      int n = n0 + j * 16 + ln;
      float bval = bo[n];
#pragma unroll
      for (int i = 0; i < 8; ++i) {
        int m = m0 + mi * 16 + i + 8 * half;
        out[m * DIM + n] = acc[mi][j][i] + bval + hs[m * DIM + n];
      }
    }
  }
}

// ---------------- Kernel 6: LayerNorm ----------------
__global__ void ln_kernel(const float* __restrict__ x, const float* __restrict__ w,
                          const float* __restrict__ b, float* __restrict__ out) {
  const int row = blockIdx.x, tid = threadIdx.x;
  const float* xr = x + row * DIM;
  float s = 0.f, s2 = 0.f;
  for (int i = tid; i < DIM; i += 256) { float v = xr[i]; s += v; s2 += v * v; }
  __shared__ float sh0[256], sh1[256];
  sh0[tid] = s; sh1[tid] = s2;
  __syncthreads();
  for (int o = 128; o > 0; o >>= 1) {
    if (tid < o) { sh0[tid] += sh0[tid + o]; sh1[tid] += sh1[tid + o]; }
    __syncthreads();
  }
  float mu = sh0[0] * (1.0f / DIM);
  float var = sh1[0] * (1.0f / DIM) - mu * mu;
  float invs = rsqrtf(var + 1e-12f);
  for (int i = tid; i < DIM; i += 256) {
    out[row * DIM + i] = (xr[i] - mu) * invs * w[i] + b[i];
  }
}

// ---------------- launch ----------------
extern "C" void kernel_launch(void* const* d_in, const int* in_sizes, int n_in,
                              void* d_out, int out_size, void* d_ws, size_t ws_size,
                              hipStream_t stream) {
  const float* hs   = (const float*)d_in[0];
  const float* cosp = (const float*)d_in[1];
  const float* sinp = (const float*)d_in[2];
  const float* Wq   = (const float*)d_in[3];
  const float* bq   = (const float*)d_in[4];
  const float* Wk   = (const float*)d_in[5];
  const float* bk   = (const float*)d_in[6];
  const float* Wv   = (const float*)d_in[7];
  const float* bv   = (const float*)d_in[8];
  const float* Wo   = (const float*)d_in[9];
  const float* bo   = (const float*)d_in[10];
  const float* lnw  = (const float*)d_in[11];
  const float* lnb  = (const float*)d_in[12];

  // workspace layout (bf16 elements then f32)
  bf16* Xbf = (bf16*)d_ws;                  // 4,194,304
  bf16* Wqb = Xbf + 4194304;                // 1,048,576 each
  bf16* Wkb = Wqb + 1048576;
  bf16* Wvb = Wkb + 1048576;
  bf16* Wob = Wvb + 1048576;
  bf16* Qb  = Wob + 1048576;                // 4,194,304 each
  bf16* Kb  = Qb + 4194304;
  bf16* Vt  = Kb + 4194304;
  bf16* Ctx = Vt + 4194304;
  float* Res = (float*)(Ctx + 4194304);     // 4,194,304 f32

  // 1) convert activations + weights to bf16 (4 elems/thread)
  cvt_kernel<<<(ROWS * DIM + 4 * 1048576) / (256 * 4), 256, 0, stream>>>(
      hs, Wq, Wk, Wv, Wo, Xbf, Wqb, Wkb, Wvb, Wob);

  // 2) QKV projections: 128 M-tiles x 16 N-strips = 2048 waves, 8 waves/block
  qkv_gemm_kernel<<<dim3(256, 3), 256, 0, stream>>>(
      Xbf, Wqb, Wkb, Wvb, bq, bk, bv, Qb, Kb, Vt);

  // 3) RoPE
  rope_kernel<<<(BATCH * HEADS * SEQ * 32) / 256, 256, 0, stream>>>(Qb, Kb, cosp, sinp);

  // 4) flash attention: 4096 waves, 4 waves/block
  attn_kernel<<<1024, 128, 0, stream>>>(Qb, Kb, Vt, Ctx);

  // 5) output projection + residual
  oproj_gemm_kernel<<<256, 256, 0, stream>>>(Ctx, Wob, bo, hs, Res);

  // 6) layernorm -> d_out
  ln_kernel<<<ROWS, 256, 0, stream>>>(Res, lnw, lnb, (float*)d_out);
}
